// GNNMoETransformerRegressor_12498354831430
// MI455X (gfx1250) — matrix operand order repaired and checked by reading, not compiled, not run on hardware
//
#include <hip/hip_runtime.h>
#include <math.h>
#include <stdint.h>

// ---------------------------------------------------------------------------
// GNN-MoE-Transformer regressor for MI455X (gfx1250, wave32, WMMA bf16).
// Double-buffered LDS pipelines, async global->LDS staging (ASYNCcnt),
// v_wmma_f32_16x16x32_bf16 with f32 accumulate everywhere that matters.
// ---------------------------------------------------------------------------

#define B_   16
#define L_   1024
#define D_   64
#define DM_  512
#define NH_  8
#define HD_  64
#define FF_  2048
#define M_   (B_*L_)        // 16384 rows

typedef __attribute__((ext_vector_type(16))) __bf16 bf16x16;
typedef __attribute__((ext_vector_type(8)))  float  f32x8;

union FragU { uint4 u[2]; bf16x16 v; };

// A-fragment (16x32, 16-bit): lane m = base row + (lane&15);
// elems 0..7  <- K = (lane<16?0:8)  + 0..7
// elems 8..15 <- K = 16 + (lane<16?0:8) + 0..7
__device__ __forceinline__ bf16x16 frag_a(const __bf16* kwin, int lane) {
  FragU f;
  const __bf16* p = kwin + ((lane & 16) ? 8 : 0);
  f.u[0] = *(const uint4*)(p);
  f.u[1] = *(const uint4*)(p + 16);
  return f.v;
}
// B-fragment (32x16, 16-bit), LDS stored n-major [n][k]:
// elems 0..7  <- K = (lane<16?0:8)  + 0..7
// elems 8..15 <- K = 16 + (lane<16?0:8) + 0..7
__device__ __forceinline__ bf16x16 frag_b(const __bf16* kwin, int lane) {
  FragU f;
  const __bf16* p = kwin + ((lane & 16) ? 8 : 0);
  f.u[0] = *(const uint4*)(p);
  f.u[1] = *(const uint4*)(p + 16);
  return f.v;
}

__device__ __forceinline__ float gelu_exact(float x) {
  return 0.5f * x * (1.0f + erff(x * 0.7071067811865476f));
}

// pack two f32 -> two bf16 in one dword (lo = a, hi = b)
__device__ __forceinline__ unsigned pack2bf(float a, float b) {
  union { __bf16 h; unsigned short u; } ua, ub;
  ua.h = (__bf16)a; ub.h = (__bf16)b;
  return ((unsigned)ub.u << 16) | (unsigned)ua.u;
}

// CDNA5 async copy: 16B global -> LDS, tracked on ASYNCcnt.
__device__ __forceinline__ void async_b128(void* lds_dst, const void* gsrc) {
  unsigned dst = (unsigned)(uintptr_t)lds_dst;   // LDS aperture: addr[31:0] = LDS offset
  unsigned long long src = (unsigned long long)(uintptr_t)gsrc;
  asm volatile("global_load_async_to_lds_b128 %0, %1, off"
               :: "v"(dst), "v"(src) : "memory");
}
__device__ __forceinline__ void wait_async0() {
  asm volatile("s_wait_asynccnt 0x0" ::: "memory");
}

// ---------------------------------------------------------------------------
// Generic 128x128-tile WMMA GEMM, double-buffered: C = epi(A @ Bw + bias)
// A: MxK row-major (ABF16 ? bf16 : f32). Bw: KxN row-major f32 (weights).
// epi: 0=bias, 1=bias+gelu, 2=bias+residual(f32), 3=bias+positional(pe)
// Requires M%128==0, N%128==0, K%32==0.
// ---------------------------------------------------------------------------
#define EPI_BIAS 0
#define EPI_GELU 1
#define EPI_RES  2
#define EPI_PE   3

template <int ABF16>
__global__ __launch_bounds__(256)
void gemm128(const void* __restrict__ Av, int lda,
             const float* __restrict__ Bw, int ldb,
             const float* __restrict__ bias,
             const float* __restrict__ res,
             const float* __restrict__ pe,
             void* __restrict__ Cv, int ldc, int outBf16,
             int Ndim, int Kdim, int epi)
{
  __shared__ __align__(16) __bf16 Alds[2][128*32];
  __shared__ __align__(16) __bf16 Blds[2][128*32];
  const int tid  = threadIdx.x;
  const int lane = tid & 31;
  const int w    = tid >> 5;
  const int wm   = w & 3;      // 4 wave-rows of 32
  const int wn   = w >> 2;     // 2 wave-cols of 64
  const int m0   = blockIdx.y * 128;
  const int n0   = blockIdx.x * 128;
  const int lhi  = lane >> 4;

  f32x8 acc[2][4];
  for (int a = 0; a < 2; ++a)
    for (int b = 0; b < 4; ++b)
      for (int e = 0; e < 8; ++e) acc[a][b][e] = 0.0f;

  const float*  Af = (const float*)Av;
  const __bf16* Ab = (const __bf16*)Av;

  // ---- prologue: stage k-tile 0 into buffer 0 ----
  if (ABF16) {
    #pragma unroll
    for (int j = 0; j < 2; ++j) {
      int idx = tid + j*256;
      int r = idx >> 2, cg = idx & 3;
      async_b128(&Alds[0][r*32 + cg*8], Ab + (size_t)(m0 + r)*lda + cg*8);
    }
  } else {
    #pragma unroll
    for (int j = 0; j < 4; ++j) {
      int idx = tid + j*256;
      int r = idx >> 3, cq = idx & 7;
      const float4 f = *(const float4*)(Af + (size_t)(m0 + r)*lda + cq*4);
      uint2 p; p.x = pack2bf(f.x, f.y); p.y = pack2bf(f.z, f.w);
      *(uint2*)(&Alds[0][r*32 + cq*4]) = p;
    }
  }
  #pragma unroll
  for (int j = 0; j < 2; ++j) {
    int idx = tid + j*256;
    int kp = idx >> 5, nq = idx & 31;
    const float4 f0 = *(const float4*)(Bw + (size_t)(2*kp    )*ldb + n0 + nq*4);
    const float4 f1 = *(const float4*)(Bw + (size_t)(2*kp + 1)*ldb + n0 + nq*4);
    *(unsigned*)(&Blds[0][(nq*4+0)*32 + 2*kp]) = pack2bf(f0.x, f1.x);
    *(unsigned*)(&Blds[0][(nq*4+1)*32 + 2*kp]) = pack2bf(f0.y, f1.y);
    *(unsigned*)(&Blds[0][(nq*4+2)*32 + 2*kp]) = pack2bf(f0.z, f1.z);
    *(unsigned*)(&Blds[0][(nq*4+3)*32 + 2*kp]) = pack2bf(f0.w, f1.w);
  }
  if (ABF16) wait_async0();
  __syncthreads();

  int cb = 0;
  for (int kt = 0; kt < Kdim; kt += 32) {
    const int  nb      = cb ^ 1;
    const bool hasNext = (kt + 32) < Kdim;

    // ---- issue next-tile global loads early (overlap with WMMA below) ----
    float4 fa[4]; float4 fb0[2], fb1[2];
    if (hasNext) {
      if (ABF16) {
        #pragma unroll
        for (int j = 0; j < 2; ++j) {
          int idx = tid + j*256;
          int r = idx >> 2, cg = idx & 3;
          async_b128(&Alds[nb][r*32 + cg*8],
                     Ab + (size_t)(m0 + r)*lda + (kt + 32) + cg*8);
        }
      } else {
        #pragma unroll
        for (int j = 0; j < 4; ++j) {
          int idx = tid + j*256;
          int r = idx >> 3, cq = idx & 7;
          fa[j] = *(const float4*)(Af + (size_t)(m0 + r)*lda + (kt + 32) + cq*4);
        }
      }
      #pragma unroll
      for (int j = 0; j < 2; ++j) {
        int idx = tid + j*256;
        int kp = idx >> 5, nq = idx & 31;
        fb0[j] = *(const float4*)(Bw + (size_t)(kt + 32 + 2*kp    )*ldb + n0 + nq*4);
        fb1[j] = *(const float4*)(Bw + (size_t)(kt + 32 + 2*kp + 1)*ldb + n0 + nq*4);
      }
    }

    // ---- compute on current buffer ----
    bf16x16 a0 = frag_a(&Alds[cb][(wm*32 +  0 + (lane&15))*32], lane);
    bf16x16 a1 = frag_a(&Alds[cb][(wm*32 + 16 + (lane&15))*32], lane);
    #pragma unroll
    for (int nt = 0; nt < 4; ++nt) {
      bf16x16 bf = frag_b(&Blds[cb][(wn*64 + nt*16 + (lane&15))*32], lane);
      acc[0][nt] = __builtin_amdgcn_wmma_f32_16x16x32_bf16(false, a0, false, bf, (short)0, acc[0][nt], false, false);
      acc[1][nt] = __builtin_amdgcn_wmma_f32_16x16x32_bf16(false, a1, false, bf, (short)0, acc[1][nt], false, false);
    }

    // ---- deferred conversion stores for next tile ----
    if (hasNext) {
      if (!ABF16) {
        #pragma unroll
        for (int j = 0; j < 4; ++j) {
          int idx = tid + j*256;
          int r = idx >> 3, cq = idx & 7;
          uint2 p; p.x = pack2bf(fa[j].x, fa[j].y); p.y = pack2bf(fa[j].z, fa[j].w);
          *(uint2*)(&Alds[nb][r*32 + cq*4]) = p;
        }
      }
      #pragma unroll
      for (int j = 0; j < 2; ++j) {
        int idx = tid + j*256;
        int kp = idx >> 5, nq = idx & 31;
        *(unsigned*)(&Blds[nb][(nq*4+0)*32 + 2*kp]) = pack2bf(fb0[j].x, fb1[j].x);
        *(unsigned*)(&Blds[nb][(nq*4+1)*32 + 2*kp]) = pack2bf(fb0[j].y, fb1[j].y);
        *(unsigned*)(&Blds[nb][(nq*4+2)*32 + 2*kp]) = pack2bf(fb0[j].z, fb1[j].z);
        *(unsigned*)(&Blds[nb][(nq*4+3)*32 + 2*kp]) = pack2bf(fb0[j].w, fb1[j].w);
      }
      if (ABF16) wait_async0();
    }
    __syncthreads();
    cb = nb;
  }

  // ---- epilogue ----
  for (int mt = 0; mt < 2; ++mt) {
    for (int nt = 0; nt < 4; ++nt) {
      int col = n0 + wn*64 + nt*16 + (lane & 15);
      float bcol = bias ? bias[col] : 0.0f;
      for (int e = 0; e < 8; ++e) {
        int row = m0 + wm*32 + mt*16 + e + 8*lhi;
        float v = acc[mt][nt][e] + bcol;
        if      (epi == EPI_GELU) v = gelu_exact(v);
        else if (epi == EPI_RES)  v += res[(size_t)row*ldc + col];
        else if (epi == EPI_PE)   v += pe[(size_t)(row & (L_-1))*Ndim + col];
        if (outBf16) ((__bf16*)Cv)[(size_t)row*ldc + col] = (__bf16)v;
        else         ((float *)Cv)[(size_t)row*ldc + col] = v;
      }
    }
  }
}

// ---------------------------------------------------------------------------
// LayerNorm over DM_=512 columns; one wave per row (16 elems/lane).
// ---------------------------------------------------------------------------
__global__ __launch_bounds__(256)
void layernorm512(const float* __restrict__ X, const float* __restrict__ g,
                  const float* __restrict__ b, float* __restrict__ Y)
{
  const int lane = threadIdx.x & 31;
  const int row  = blockIdx.x * 8 + (threadIdx.x >> 5);
  const float* x = X + (size_t)row * DM_;
  float v[16];
  float s = 0.f;
  for (int i = 0; i < 16; ++i) { v[i] = x[lane + i*32]; s += v[i]; }
  for (int m = 16; m >= 1; m >>= 1) s += __shfl_xor(s, m, 32);
  float mu = s * (1.0f/DM_);
  float q = 0.f;
  for (int i = 0; i < 16; ++i) { float d = v[i]-mu; q += d*d; }
  for (int m = 16; m >= 1; m >>= 1) q += __shfl_xor(q, m, 32);
  float inv = rsqrtf(q*(1.0f/DM_) + 1e-5f);
  float* y = Y + (size_t)row*DM_;
  for (int i = 0; i < 16; ++i) {
    int c = lane + i*32;
    y[c] = (v[i]-mu)*inv*g[c] + b[c];
  }
}

// ---------------------------------------------------------------------------
// Flash-style attention, double-buffered K/V. Block = one (b,h, 128-q tile);
// 8 waves, each wave owns 16 q-rows. QP buffer: Q staging first, then per-wave
// P tiles (wave w's P region == wave w's Q rows; no cross-wave overlap).
// qkv layout: [b*L+l][1536] bf16 (q:0..511, k:512..1023, v:1024..1535).
// ---------------------------------------------------------------------------
__global__ __launch_bounds__(256)
void attention(const void* __restrict__ qkv_, void* __restrict__ O_)
{
  __shared__ __align__(16) __bf16 QP[128*64];     // Q staging, then P tiles
  __shared__ __align__(16) __bf16 Ks[2][64*64];   // [key][d]  (n-major for QK^T)
  __shared__ __align__(16) __bf16 Vs[2][64*64];   // [d][key]  (n-major for P*V)
  const __bf16* qkv = (const __bf16*)qkv_;
  __bf16* O = (__bf16*)O_;
  const int tid = threadIdx.x, lane = tid & 31, w = tid >> 5;
  const int bh = blockIdx.x;            // 0..127
  const int b  = bh >> 3, hh = bh & 7;
  const int q0 = blockIdx.y * 128;
  const size_t rowbase = (size_t)b * L_;
  const int lhi = lane >> 4;

  // stage Q tile (128x64) via async copy
  #pragma unroll
  for (int j = 0; j < 4; ++j) {
    int idx = tid + j*256;
    int r = idx >> 3, dg = idx & 7;
    async_b128(&QP[r*64 + dg*8],
               qkv + (rowbase + q0 + r)*1536 + hh*64 + dg*8);
  }
  wait_async0();
  __syncthreads();
  bf16x16 aq0 = frag_a(&QP[(w*16 + (lane&15))*64 +  0], lane);
  bf16x16 aq1 = frag_a(&QP[(w*16 + (lane&15))*64 + 32], lane);
  __bf16* Pw = &QP[w*16*64];            // wave-private P tile [16][64]

  float mstate[8], lstate[8];
  f32x8 oacc[4];
  for (int e = 0; e < 8; ++e) { mstate[e] = -1e30f; lstate[e] = 0.f; }
  for (int nt = 0; nt < 4; ++nt) for (int e = 0; e < 8; ++e) oacc[nt][e] = 0.f;

  // prologue: stage key-tile 0 into buffer 0 (K async, V transposed)
  #pragma unroll
  for (int j = 0; j < 2; ++j) {
    int idx = tid + j*256;
    int r = idx >> 3, dg = idx & 7;
    async_b128(&Ks[0][r*64 + dg*8],
               qkv + (rowbase + r)*1536 + 512 + hh*64 + dg*8);
    uint4 uv = *(const uint4*)(qkv + (rowbase + r)*1536 + 1024 + hh*64 + dg*8);
    const __bf16* pv = (const __bf16*)&uv;
    for (int q = 0; q < 8; ++q) Vs[0][(dg*8+q)*64 + r] = pv[q];
  }
  wait_async0();
  __syncthreads();

  int cb = 0;
  for (int kt0 = 0; kt0 < L_; kt0 += 64) {
    const int  nb      = cb ^ 1;
    const bool hasNext = (kt0 + 64) < L_;

    // ---- issue next key-tile loads early ----
    uint4 vreg[2];
    if (hasNext) {
      #pragma unroll
      for (int j = 0; j < 2; ++j) {
        int idx = tid + j*256;
        int r = idx >> 3, dg = idx & 7;
        async_b128(&Ks[nb][r*64 + dg*8],
                   qkv + (rowbase + kt0 + 64 + r)*1536 + 512 + hh*64 + dg*8);
        vreg[j] = *(const uint4*)(qkv + (rowbase + kt0 + 64 + r)*1536 + 1024 + hh*64 + dg*8);
      }
    }

    // ---- S = (Q K^T) * 1/sqrt(64) ----
    f32x8 sac[4];
    #pragma unroll
    for (int nt = 0; nt < 4; ++nt) {
      for (int e = 0; e < 8; ++e) sac[nt][e] = 0.f;
      bf16x16 bk0 = frag_b(&Ks[cb][(nt*16 + (lane&15))*64 +  0], lane);
      bf16x16 bk1 = frag_b(&Ks[cb][(nt*16 + (lane&15))*64 + 32], lane);
      sac[nt] = __builtin_amdgcn_wmma_f32_16x16x32_bf16(false, aq0, false, bk0, (short)0, sac[nt], false, false);
      sac[nt] = __builtin_amdgcn_wmma_f32_16x16x32_bf16(false, aq1, false, bk1, (short)0, sac[nt], false, false);
    }

    // ---- online softmax per q-row; row e lives in 16-lane half-waves ----
    #pragma unroll
    for (int e = 0; e < 8; ++e) {
      float s0 = sac[0][e]*0.125f, s1 = sac[1][e]*0.125f;
      float s2 = sac[2][e]*0.125f, s3 = sac[3][e]*0.125f;
      float mx = fmaxf(fmaxf(s0,s1), fmaxf(s2,s3));
      for (int m = 8; m >= 1; m >>= 1) mx = fmaxf(mx, __shfl_xor(mx, m, 32));
      float mnew  = fmaxf(mstate[e], mx);
      float alpha = __expf(mstate[e] - mnew);
      float p0 = __expf(s0 - mnew), p1 = __expf(s1 - mnew);
      float p2 = __expf(s2 - mnew), p3 = __expf(s3 - mnew);
      float rs = p0 + p1 + p2 + p3;
      for (int m = 8; m >= 1; m >>= 1) rs += __shfl_xor(rs, m, 32);
      mstate[e] = mnew;
      lstate[e] = lstate[e]*alpha + rs;
      for (int nt = 0; nt < 4; ++nt) oacc[nt][e] *= alpha;
      __bf16* pr = &Pw[(e + 8*lhi)*64 + (lane&15)];
      pr[0] = (__bf16)p0; pr[16] = (__bf16)p1; pr[32] = (__bf16)p2; pr[48] = (__bf16)p3;
    }
    __syncthreads();   // make P LDS writes visible across lanes

    // ---- O += P * V  (K = 64 keys -> 2 k-steps; N = 64 dims -> 4 n-tiles) ----
    #pragma unroll
    for (int s = 0; s < 2; ++s) {
      bf16x16 ap = frag_a(&Pw[(lane&15)*64 + s*32], lane);
      #pragma unroll
      for (int nt = 0; nt < 4; ++nt) {
        bf16x16 bv = frag_b(&Vs[cb][(nt*16 + (lane&15))*64 + s*32], lane);
        oacc[nt] = __builtin_amdgcn_wmma_f32_16x16x32_bf16(false, ap, false, bv, (short)0, oacc[nt], false, false);
      }
    }

    // ---- deferred V transpose stores + async wait ----
    if (hasNext) {
      #pragma unroll
      for (int j = 0; j < 2; ++j) {
        int idx = tid + j*256;
        int r = idx >> 3, dg = idx & 7;
        const __bf16* pv = (const __bf16*)&vreg[j];
        for (int q = 0; q < 8; ++q) Vs[nb][(dg*8+q)*64 + r] = pv[q];
      }
      wait_async0();
    }
    __syncthreads();
    cb = nb;
  }

  for (int e = 0; e < 8; ++e) {
    float invl = 1.0f / lstate[e];
    int l = q0 + w*16 + e + 8*lhi;
    __bf16* orow = O + (rowbase + l)*DM_ + hh*64;
    for (int nt = 0; nt < 4; ++nt)
      orow[nt*16 + (lane&15)] = (__bf16)(oacc[nt][e] * invl);
  }
}

// ---------------------------------------------------------------------------
// GEMV: out[row] = dot(A[row,:], wv) + bias[0].  One wave per row.
// ---------------------------------------------------------------------------
__global__ __launch_bounds__(256)
void gemv_rows(const void* __restrict__ Av, int aBf16, int Kdim,
               const float* __restrict__ wv, const float* __restrict__ bias,
               float* __restrict__ out)
{
  const int lane = threadIdx.x & 31;
  const int row  = blockIdx.x*8 + (threadIdx.x >> 5);
  float s = 0.f;
  if (aBf16) {
    const __bf16* a = (const __bf16*)Av + (size_t)row*Kdim;
    for (int c = lane; c < Kdim; c += 32) s += (float)a[c]*wv[c];
  } else {
    const float* a = (const float*)Av + (size_t)row*Kdim;
    for (int c = lane; c < Kdim; c += 32) s += a[c]*wv[c];
  }
  for (int m = 16; m >= 1; m >>= 1) s += __shfl_xor(s, m, 32);
  if (lane == 0) out[row] = s + bias[0];
}

// ---------------------------------------------------------------------------
// Trend expert, algebraically folded: g[b][t] = sum_c (conv(h) + cb)*outw.
// ---------------------------------------------------------------------------
__global__ __launch_bounds__(256)
void trend_g(const float* __restrict__ h, const float* __restrict__ convw,
             const float* __restrict__ convb, const float* __restrict__ outw,
             float* __restrict__ g)
{
  const int lane = threadIdx.x & 31;
  const int e = blockIdx.x*8 + (threadIdx.x >> 5);  // 0..4095
  const int b = e >> 8, t = e & 255;
  const float* hb = h + ((size_t)b*L_ + t*4)*DM_;
  float s = 0.f;
  for (int c = lane; c < DM_; c += 32) {
    float d = convb[c];
    for (int ss = 0; ss < 4; ++ss) d += hb[ss*DM_ + c] * convw[c*4 + ss];
    s += d * outw[c];
  }
  for (int m = 16; m >= 1; m >>= 1) s += __shfl_xor(s, m, 32);
  if (lane == 0) g[b*256 + t] = s;
}

// ---------------------------------------------------------------------------
// Final mixture: y = pred + pbar[1]*interp(g) + pbar[3]*value.
// ---------------------------------------------------------------------------
__global__ __launch_bounds__(256)
void mix_out(const float* __restrict__ pred, const float* __restrict__ val,
             const float* __restrict__ g, const float* __restrict__ pbar,
             const float* __restrict__ tb, float* __restrict__ y)
{
  int i = blockIdx.x*256 + threadIdx.x;
  int b = i >> 10, l = i & 1023;
  float pos = (l + 0.5f)*0.25f - 0.5f;
  pos = fminf(fmaxf(pos, 0.f), 255.f);
  int i0 = (int)floorf(pos);
  int i1 = (i0 + 1 > 255) ? 255 : i0 + 1;
  float wg = pos - (float)i0;
  float tr = (1.f - wg)*g[b*256 + i0] + wg*g[b*256 + i1] + tb[0];
  y[i] = pred[i] + pbar[1]*tr + pbar[3]*val[i];
}

// ---------------------------------------------------------------------------
// Pearson column stats over flattened (B*L, D) matrix.
// ---------------------------------------------------------------------------
__global__ __launch_bounds__(256)
void col_stats(const float* __restrict__ x, float* __restrict__ cmean,
               float* __restrict__ cinv)
{
  __shared__ float sh[256];
  const int d = blockIdx.x;
  float s = 0.f;
  for (int r = threadIdx.x; r < M_; r += 256) s += x[(size_t)r*D_ + d];
  sh[threadIdx.x] = s; __syncthreads();
  for (int st = 128; st >= 1; st >>= 1) {
    if (threadIdx.x < st) sh[threadIdx.x] += sh[threadIdx.x + st];
    __syncthreads();
  }
  float mu = sh[0] * (1.0f/M_);
  __syncthreads();
  float q = 0.f;
  for (int r = threadIdx.x; r < M_; r += 256) { float dv = x[(size_t)r*D_ + d]-mu; q += dv*dv; }
  sh[threadIdx.x] = q; __syncthreads();
  for (int st = 128; st >= 1; st >>= 1) {
    if (threadIdx.x < st) sh[threadIdx.x] += sh[threadIdx.x + st];
    __syncthreads();
  }
  if (threadIdx.x == 0) {
    cmean[d] = mu;
    float sd = sqrtf(sh[0]/(float)(M_-1));
    cinv[d] = 1.0f / fmaxf(sd, 1e-6f);
  }
}

__global__ __launch_bounds__(256)
void corr_mask(const float* __restrict__ x, const float* __restrict__ cmean,
               const float* __restrict__ cinv, float* __restrict__ mask)
{
  const int lane = threadIdx.x & 31;
  const int p = blockIdx.x*8 + (threadIdx.x >> 5);  // 0..4095
  const int i = p >> 6, j = p & 63;
  float mi = cmean[i], mj = cmean[j];
  float s = 0.f;
  for (int r = lane; r < M_; r += 32)
    s += (x[(size_t)r*D_ + i]-mi)*(x[(size_t)r*D_ + j]-mj);
  for (int m = 16; m >= 1; m >>= 1) s += __shfl_xor(s, m, 32);
  if (lane == 0) {
    float corr = s * cinv[i] * cinv[j] * (1.0f/(float)(M_-1));
    bool on = (i == j) || (fabsf(corr) >= 0.3f);
    mask[i*64 + j] = on ? 1.0f : 0.0f;
  }
}

// per-variable node features: mean over b of (mean_l, std_l(ddof=1))
__global__ __launch_bounds__(256)
void node_feats(const float* __restrict__ x, float* __restrict__ n0)
{
  __shared__ float sh[256];
  const int d = blockIdx.x;
  float am = 0.f, as = 0.f;
  for (int b = 0; b < B_; ++b) {
    float s = 0.f;
    for (int l = threadIdx.x; l < L_; l += 256) s += x[((size_t)b*L_ + l)*D_ + d];
    sh[threadIdx.x] = s; __syncthreads();
    for (int st = 128; st >= 1; st >>= 1) {
      if (threadIdx.x < st) sh[threadIdx.x] += sh[threadIdx.x + st];
      __syncthreads();
    }
    float mu = sh[0]*(1.0f/L_);
    __syncthreads();
    float q = 0.f;
    for (int l = threadIdx.x; l < L_; l += 256) { float dv = x[((size_t)b*L_ + l)*D_ + d]-mu; q += dv*dv; }
    sh[threadIdx.x] = q; __syncthreads();
    for (int st = 128; st >= 1; st >>= 1) {
      if (threadIdx.x < st) sh[threadIdx.x] += sh[threadIdx.x + st];
      __syncthreads();
    }
    if (threadIdx.x == 0) { am += mu; as += sqrtf(sh[0]/(float)(L_-1)); }
    __syncthreads();
  }
  if (threadIdx.x == 0) { n0[d*2+0] = am*(1.0f/B_); n0[d*2+1] = as*(1.0f/B_); }
}

// masked-GAT attention sub-step: H = gelu(softmax(leaky(adst_i+asrc_j), mask) @ HW + b)
__device__ void gat_attend_out(const float* __restrict__ mask,
                               const float* asv, const float* adv, const float* bb,
                               const float* HW, float* Hout,
                               float* ATT, float* asrc, float* adst)
{
  const int tid = threadIdx.x;
  if (tid < 128) {
    int i = tid & 63;
    const float* av = (tid < 64) ? asv : adv;
    float s = 0.f;
    for (int c = 0; c < 64; ++c) s += HW[i*64 + c]*av[c];
    if (tid < 64) asrc[i] = s; else adst[i] = s;
  }
  __syncthreads();
  if (tid < 64) {
    int i = tid;
    float sc[64];
    float mx = -1e30f;
    for (int j = 0; j < 64; ++j) {
      float s = adst[i] + asrc[j];
      s = (s > 0.f) ? s : 0.2f*s;
      if (mask[i*64 + j] == 0.f) s = -1e30f;
      sc[j] = s; mx = fmaxf(mx, s);
    }
    float den = 0.f;
    for (int j = 0; j < 64; ++j) {
      float e = (sc[j] <= -1e29f) ? 0.f : __expf(sc[j]-mx);
      sc[j] = e; den += e;
    }
    float inv = 1.0f/den;
    for (int j = 0; j < 64; ++j) ATT[i*64 + j] = sc[j]*inv;
  }
  __syncthreads();
  for (int idx = tid; idx < 4096; idx += 256) {
    int i = idx >> 6, co = idx & 63;
    float s = bb[co];
    for (int j = 0; j < 64; ++j) s += ATT[i*64 + j]*HW[j*64 + co];
    Hout[idx] = gelu_exact(s);
  }
  __syncthreads();
}

__global__ __launch_bounds__(256)
void gat_gate(const float* __restrict__ n0, const float* __restrict__ mask,
              const float* w0, const float* as0, const float* ad0, const float* b0,
              const float* w1, const float* as1, const float* ad1, const float* b1,
              const float* l1w, const float* l1b, const float* l2w, const float* l2b,
              float* __restrict__ pbar)
{
  __shared__ float H[64*64];
  __shared__ float HW[64*64];
  __shared__ float ATT[64*64];
  __shared__ float asrc[64], adst[64];
  __shared__ float ps[64*4];
  const int tid = threadIdx.x;

  // layer 1: HW = n0(64x2) @ w0(2x64)
  for (int idx = tid; idx < 4096; idx += 256) {
    int i = idx >> 6, co = idx & 63;
    HW[idx] = n0[i*2+0]*w0[0*64 + co] + n0[i*2+1]*w0[1*64 + co];
  }
  __syncthreads();
  gat_attend_out(mask, as0, ad0, b0, HW, H, ATT, asrc, adst);

  // layer 2: HW = H @ w1(64x64)
  for (int idx = tid; idx < 4096; idx += 256) {
    int i = idx >> 6, co = idx & 63;
    float s = 0.f;
    for (int c = 0; c < 64; ++c) s += H[i*64 + c]*w1[c*64 + co];
    HW[idx] = s;
  }
  __syncthreads();
  gat_attend_out(mask, as1, ad1, b1, HW, H, ATT, asrc, adst);

  // gate MLP: HW = gelu(H @ l1w + l1b)
  for (int idx = tid; idx < 4096; idx += 256) {
    int i = idx >> 6, co = idx & 63;
    float s = l1b[co];
    for (int c = 0; c < 64; ++c) s += H[i*64 + c]*l1w[c*64 + co];
    HW[idx] = gelu_exact(s);
  }
  __syncthreads();
  if (tid < 64) {
    int i = tid;
    float lg[4]; float mx = -1e30f;
    for (int k = 0; k < 4; ++k) {
      float s = l2b[k];
      for (int c = 0; c < 64; ++c) s += HW[i*64 + c]*l2w[c*4 + k];
      lg[k] = s; mx = fmaxf(mx, s);
    }
    float den = 0.f;
    for (int k = 0; k < 4; ++k) { lg[k] = __expf(lg[k]-mx); den += lg[k]; }
    for (int k = 0; k < 4; ++k) ps[i*4 + k] = lg[k]/den;
  }
  __syncthreads();
  if (tid < 4) {
    float s = 0.f;
    for (int i = 0; i < 64; ++i) s += ps[i*4 + tid];
    pbar[tid] = s*(1.0f/64.0f);
  }
}

// ---------------------------------------------------------------------------
// Host launcher.  Input ordering assumes depth-first insertion-order
// flattening of setup_inputs(): x, gat0{w,a_src,a_dst,b}, gat1{...},
// gate{l1.w,l1.b,l2.w,l2.b}, in_proj{w,b}, pe,
// 4x layer{wqkv.w,wqkv.b,wo.w,wo.b,ln1g,ln1b,ln2g,ln2b,ff1.w,ff1.b,ff2.w,ff2.b},
// head{w,b}, trend{convw,convb,out.w,out.b}, value{l1.w,l1.b,l2.w,l2.b,l3.w,l3.b}
// ---------------------------------------------------------------------------
extern "C" void kernel_launch(void* const* d_in, const int* in_sizes, int n_in,
                              void* d_out, int out_size, void* d_ws, size_t ws_size,
                              hipStream_t stream)
{
  (void)in_sizes; (void)n_in; (void)out_size; (void)ws_size;
  auto F = [&](int i) { return (const float*)d_in[i]; };
  const float* x = F(0);

  // workspace carve-out
  char* wp = (char*)d_ws;
  auto alloc = [&](size_t bytes) {
    void* p = (void*)wp;
    wp += (bytes + 255) & ~(size_t)255;
    return p;
  };
  float* pbar   = (float*)alloc(4*sizeof(float));
  float* cmean  = (float*)alloc(64*sizeof(float));
  float* cinv   = (float*)alloc(64*sizeof(float));
  float* maskb  = (float*)alloc(64*64*sizeof(float));
  float* n0buf  = (float*)alloc(128*sizeof(float));
  float* h      = (float*)alloc((size_t)M_*DM_*4);     // residual stream f32
  float* t1     = (float*)alloc((size_t)M_*DM_*4);     // LN output f32
  void*  qkv    = alloc((size_t)M_*1536*2);            // bf16
  void*  obuf   = alloc((size_t)M_*DM_*2);             // bf16
  void*  fbuf   = alloc((size_t)M_*FF_*2);             // bf16 FFN hidden
  void*  v1     = alloc((size_t)M_*128*2);             // bf16
  void*  v2     = alloc((size_t)M_*128*2);             // bf16
  float* pred   = (float*)alloc((size_t)M_*4);
  float* val    = (float*)alloc((size_t)M_*4);
  float* gbuf   = (float*)alloc((size_t)B_*256*4);

  // ---- irregular graph/GNN/gate path (tiny) ----
  col_stats <<<64,  256, 0, stream>>>(x, cmean, cinv);
  corr_mask <<<512, 256, 0, stream>>>(x, cmean, cinv, maskb);
  node_feats<<<64,  256, 0, stream>>>(x, n0buf);
  gat_gate  <<<1,   256, 0, stream>>>(n0buf, maskb,
                                      F(1), F(2), F(3), F(4),
                                      F(5), F(6), F(7), F(8),
                                      F(9), F(10), F(11), F(12),
                                      pbar);

  // ---- backbone: in_proj + positional encoding ----
  gemm128<0><<<dim3(DM_/128, M_/128), 256, 0, stream>>>(
      x, D_, F(13), DM_, F(14), nullptr, F(15),
      h, DM_, 0, DM_, D_, EPI_PE);

  for (int l = 0; l < 4; ++l) {
    int base = 16 + 12*l;
    const float* wqkvw = F(base+0);  const float* wqkvb = F(base+1);
    const float* wow   = F(base+2);  const float* wob   = F(base+3);
    const float* ln1g  = F(base+4);  const float* ln1b  = F(base+5);
    const float* ln2g  = F(base+6);  const float* ln2b  = F(base+7);
    const float* ff1w  = F(base+8);  const float* ff1b  = F(base+9);
    const float* ff2w  = F(base+10); const float* ff2b  = F(base+11);

    layernorm512<<<M_/8, 256, 0, stream>>>(h, ln1g, ln1b, t1);
    gemm128<0><<<dim3(1536/128, M_/128), 256, 0, stream>>>(
        t1, DM_, wqkvw, 1536, wqkvb, nullptr, nullptr,
        qkv, 1536, 1, 1536, DM_, EPI_BIAS);
    attention<<<dim3(B_*NH_, L_/128), 256, 0, stream>>>(qkv, obuf);
    gemm128<1><<<dim3(DM_/128, M_/128), 256, 0, stream>>>(
        obuf, DM_, wow, DM_, wob, h, nullptr,
        h, DM_, 0, DM_, DM_, EPI_RES);

    layernorm512<<<M_/8, 256, 0, stream>>>(h, ln2g, ln2b, t1);
    gemm128<0><<<dim3(FF_/128, M_/128), 256, 0, stream>>>(
        t1, DM_, ff1w, FF_, ff1b, nullptr, nullptr,
        fbuf, FF_, 1, FF_, DM_, EPI_GELU);
    gemm128<1><<<dim3(DM_/128, M_/128), 256, 0, stream>>>(
        fbuf, FF_, ff2w, DM_, ff2b, h, nullptr,
        h, DM_, 0, DM_, FF_, EPI_RES);
  }

  // ---- heads / experts ----
  gemv_rows<<<M_/8, 256, 0, stream>>>(h, 0, DM_, F(64), F(65), pred);

  gemm128<0><<<dim3(1, M_/128), 256, 0, stream>>>(
      h, DM_, F(70), 128, F(71), nullptr, nullptr,
      v1, 128, 1, 128, DM_, EPI_GELU);
  gemm128<1><<<dim3(1, M_/128), 256, 0, stream>>>(
      v1, 128, F(72), 128, F(73), nullptr, nullptr,
      v2, 128, 1, 128, 128, EPI_GELU);
  gemv_rows<<<M_/8, 256, 0, stream>>>(v2, 1, 128, F(74), F(75), val);

  trend_g<<<(B_*256)/8, 256, 0, stream>>>(h, F(66), F(67), F(68), gbuf);

  mix_out<<<M_/256, 256, 0, stream>>>(pred, val, gbuf, pbar, F(69), (float*)d_out);
}
